// Djpegnet_27212912787694
// MI455X (gfx1250) — matrix operand order repaired and checked by reading, not compiled, not run on hardware
//
#include <hip/hip_runtime.h>
#include <hip/hip_bf16.h>

// ---------------------------------------------------------------------------
// DJPEGNet for MI455X (gfx1250, wave32).  v4:
//  - conv weights pre-converted f32->f16 once; conv A-tiles then staged with
//    GLOBAL_LOAD_ASYNC_TO_LDS_B128 (K%32==0 layers), B-tiles im2col-gathered.
//  - conv block tile M=64 x N=64, two accumulators/wave (A-fragment reuse,
//    2 v_wmma per K-step between one barrier pair).
//  - DCT kernel stages its 8x256 f32 image slab with TENSOR_LOAD_TO_LDS.
//  - fragments fetched as 16B ds_load_b128 in contiguous k order (k = 16h + s).
// ---------------------------------------------------------------------------

typedef __attribute__((ext_vector_type(16))) _Float16 v16h;
typedef __attribute__((ext_vector_type(8)))  _Float16 v8h;
typedef __attribute__((ext_vector_type(8)))  float    v8f;
typedef __attribute__((ext_vector_type(4)))  unsigned uint4v;
typedef __attribute__((ext_vector_type(8)))  int      int8v;
typedef __attribute__((ext_vector_type(4)))  int      int4v;

#if __has_builtin(__builtin_amdgcn_global_load_async_to_lds_b128) && \
    __has_builtin(__builtin_amdgcn_s_wait_asynccnt)
#define HAS_ASYNC_LDS 1
#else
#define HAS_ASYNC_LDS 0
#endif
#if __has_builtin(__builtin_amdgcn_tensor_load_to_lds) && \
    __has_builtin(__builtin_amdgcn_s_wait_tensorcnt)
#define HAS_TDM 1
#else
#define HAS_TDM 0
#endif

// Builtin signature (verified via hipcc diagnostic): (int4 __device__*,
// int4 LDS*, imm offset, imm cpol).
#define GAS_I4(p) ((__attribute__((address_space(1))) int4v*)(p))
#define LAS_I4(p) ((__attribute__((address_space(3))) int4v*)(p))
#define AS3P(p)   ((__attribute__((address_space(3))) void*)(p))

// Build a v16h WMMA fragment from two aligned 8-half LDS chunks (2x ds_load_b128)
__device__ __forceinline__ v16h frag16(const _Float16* p, int off_lo, int off_hi) {
  v8h lo = *(const v8h*)(p + off_lo);
  v8h hi = *(const v8h*)(p + off_hi);
  return __builtin_shufflevector(lo, hi, 0, 1, 2, 3, 4, 5, 6, 7,
                                 8, 9, 10, 11, 12, 13, 14, 15);
}

// ---------------------------------------------------------------------------
// Kernel 0: one-shot f32 -> f16 conversion (conv weights, 2.3 MB total).
// ---------------------------------------------------------------------------
__global__ void cvt_f16_kernel(const float* __restrict__ src,
                               _Float16* __restrict__ dst, int n) {
  int i = blockIdx.x * 256 + threadIdx.x;
  if (i < n) dst[i] = (_Float16)src[i];
}

// ---------------------------------------------------------------------------
// Kernel 1: 8x8 block DCT as GEMM.  Block = (image b, block-row by):
// stages 8 image rows (8x256 f32) via TDM, basis as f16 in LDS, computes
// M=64 x N=32 outputs with 8 waves.  grid = 128*32 = 4096 blocks.
// ---------------------------------------------------------------------------
__global__ void dct_wmma_kernel(const float* __restrict__ x,
                                const float* __restrict__ basis,
                                float* __restrict__ dct) {
  __shared__ __align__(16) float    xrow[8 * 256];
  __shared__ __align__(16) _Float16 basF[64 * 72];
  int tid = threadIdx.x, lane = tid & 31, wv = tid >> 5;
  int b  = blockIdx.x >> 5;
  int by = blockIdx.x & 31;
  const float* xsrc = x + (b * 256 + by * 8) * 256;

#if HAS_TDM
  if (wv == 0) {
    unsigned lds_base = (unsigned)(unsigned long long)AS3P(&xrow[0]);
    unsigned long long ga = (unsigned long long)xsrc;
    uint4v g0;
    g0[0] = 1u;                                        // count=1
    g0[1] = lds_base;                                  // lds_addr
    g0[2] = (unsigned)(ga & 0xFFFFFFFFu);              // global_addr lo
    g0[3] = (unsigned)((ga >> 32) & 0x01FFFFFFu) | (2u << 30);  // hi | type=2
    int8v g1 = {};
    g1[0] = (2 << 16);                 // data_size code 2 = 4 bytes
    g1[1] = (int)(256u << 16);         // tensor_dim0 = 256
    g1[2] = (int)(8u << 16);           // tensor_dim1 = 8
    g1[3] = (int)(256u << 16);         // tile_dim0 = 256
    g1[4] = 8;                         // tile_dim1 = 8
    g1[5] = 256;                       // tensor_dim0_stride = 256
    int4v z4 = {};
#if __clang_major__ >= 23
    int8v z8 = {};
    __builtin_amdgcn_tensor_load_to_lds(g0, g1, z4, z4, z8, 0);
#else
    __builtin_amdgcn_tensor_load_to_lds(g0, g1, z4, z4, 0);
#endif
  }
#elif HAS_ASYNC_LDS
  for (int i = tid; i < 512; i += 256)
    __builtin_amdgcn_global_load_async_to_lds_b128(GAS_I4(xsrc + i * 4),
                                                   LAS_I4(&xrow[i * 4]), 0, 0);
#else
  for (int i = tid; i < 2048; i += 256) xrow[i] = xsrc[i];
#endif

  for (int i = tid; i < 4096; i += 256) {       // stage DCT basis f32->f16
    int m = i >> 6, k = i & 63;
    basF[m * 72 + k] = (_Float16)basis[i];
  }

#if HAS_TDM
  if (wv == 0) __builtin_amdgcn_s_wait_tensorcnt(0);
#elif HAS_ASYNC_LDS
  __builtin_amdgcn_s_wait_asynccnt(0);
#endif
  __syncthreads();

  int arow  = ((wv >> 1) * 16) + (lane & 15);
  int kc    = (lane & 16) ? 16 : 0;
  int bx    = ((wv & 1) * 16) + (lane & 15);
  int kboff = (lane & 16) ? 16 : 0;
  v8f acc = {};
  for (int kt = 0; kt < 2; ++kt) {
    v16h A = frag16(&basF[arow * 72 + kt * 32], kc, kc + 8);
    v16h Bm;
    for (int h = 0; h < 16; ++h) {
      int k = kt * 32 + kboff + h;
      int dy = k >> 3, dx = k & 7;
      Bm[h] = (_Float16)xrow[dy * 256 + bx * 8 + dx];
    }
    acc = __builtin_amdgcn_wmma_f32_16x16x32_f16(false, A, false, Bm,
                                                 (short)0, acc, false, false);
  }
  for (int r = 0; r < 8; ++r) {
    int om = ((wv >> 1) * 16) + r + ((lane & 16) ? 8 : 0);
    dct[(b * 64 + om) * 1024 + by * 32 + bx] = acc[r];
  }
}

// ---------------------------------------------------------------------------
// Kernel 2: soft histogram (gamma=1e6 -> hard histogram of floor(d+60)).
// ---------------------------------------------------------------------------
__global__ void hist_kernel(const float* __restrict__ dct,
                            _Float16* __restrict__ feat) {
  __shared__ float hist[120];
  int t = threadIdx.x;
  if (t < 120) hist[t] = 0.0f;
  __syncthreads();
  int b = blockIdx.x >> 6, c = blockIdx.x & 63;
  const float* p = dct + (b * 64 + c) * 1024;
  for (int i = t; i < 1024; i += 256) {
    int bin = (int)floorf(p[i] + 60.0f);
    if (bin >= 0 && bin < 120) atomicAdd(&hist[bin], 1.0f);
  }
  __syncthreads();
  if (t < 120)
    feat[(b * 120 + t) * 64 + c] = (_Float16)(hist[t] * (1.0f / 1024.0f));
}

// ---------------------------------------------------------------------------
// Kernel 3: 5x5 pad-2 conv + BN + ReLU, LDS-tiled implicit GEMM.
// Block tile M=64 x N=64, 8 waves, 2 accumulators/wave (shared A fragment).
// A tile staged async from pre-converted f16 weights when K%32==0.
// grid = (O/64) * (128*H*W/64) blocks.
// ---------------------------------------------------------------------------
__global__ void conv_bn_relu_wmma(const _Float16* __restrict__ in,
                                  const _Float16* __restrict__ w16,
                                  const float* __restrict__ bias,
                                  const float* __restrict__ bns,
                                  const float* __restrict__ bnb,
                                  const float* __restrict__ bnm,
                                  const float* __restrict__ bnv,
                                  _Float16* __restrict__ out,
                                  int C, int O, int H, int W) {
  __shared__ __align__(16) _Float16 Af[64 * 40];
  __shared__ __align__(16) _Float16 Bf[64 * 40];
  int tid = threadIdx.x, lane = tid & 31, wv = tid >> 5;
  int nblocks = (128 * H * W) >> 6;
  int mb = blockIdx.x / nblocks, nb = blockIdx.x % nblocks;
  int m0 = mb * 64, n0 = nb * 64;
  int K = C * 25, ksteps = (K + 31) >> 5;
  int arow  = ((wv >> 1) * 16) + (lane & 15);
  int kc    = (lane & 16) ? 16 : 0;
  int nsub  = (wv & 1) * 32;                    // this wave's 32-wide N strip
  int brow0 = nsub + (lane & 15);
  int brow1 = nsub + 16 + (lane & 15);
  int kboff = (lane & 16) ? 16 : 0;
#if HAS_ASYNC_LDS
  bool kAligned = (K & 31) == 0;                // conv1b/2a/3a; conv1a K=25
#endif
  v8f acc0 = {}, acc1 = {};
  for (int ks = 0; ks < ksteps; ++ks) {
    int k0 = ks << 5;
#if HAS_ASYNC_LDS
    if (kAligned) {                             // A: 64 rows x 4 b128 chunks
      int m = tid >> 2, q = tid & 3;
      const _Float16* gp = w16 + (size_t)(m0 + m) * K + k0 + q * 8;
      __builtin_amdgcn_global_load_async_to_lds_b128(GAS_I4(gp),
                                                     LAS_I4(&Af[m * 40 + q * 8]),
                                                     0, 0);
    } else
#endif
    {
      for (int i = tid; i < 2048; i += 256) {
        int m = i >> 5, kk = i & 31, k = k0 + kk;
        Af[m * 40 + kk] = (k < K) ? w16[(m0 + m) * K + k] : (_Float16)0.0f;
      }
    }
    for (int i = tid; i < 2048; i += 256) {     // B: im2col gather, 64 rows
      int n = i >> 5, kk = i & 31, k = k0 + kk;
      _Float16 bv = (_Float16)0.0f;
      if (k < K) {
        int gn = n0 + n;
        int xc = gn % W; int t2 = gn / W;
        int y = t2 % H;  int bb = t2 / H;
        int c = k / 25,  r = k % 25;
        int iy = y + r / 5 - 2, ix = xc + r % 5 - 2;
        if (iy >= 0 && iy < H && ix >= 0 && ix < W)
          bv = in[((bb * C + c) * H + iy) * W + ix];
      }
      Bf[n * 40 + kk] = bv;
    }
#if HAS_ASYNC_LDS
    if (kAligned) __builtin_amdgcn_s_wait_asynccnt(0);
#endif
    __syncthreads();
    v16h A  = frag16(&Af[arow * 40], kc, kc + 8);
    v16h B0 = frag16(&Bf[brow0 * 40 + kboff], 0, 8);
    v16h B1 = frag16(&Bf[brow1 * 40 + kboff], 0, 8);
    acc0 = __builtin_amdgcn_wmma_f32_16x16x32_f16(false, A, false, B0,
                                                  (short)0, acc0, false, false);
    acc1 = __builtin_amdgcn_wmma_f32_16x16x32_f16(false, A, false, B1,
                                                  (short)0, acc1, false, false);
    __syncthreads();
  }
  for (int r = 0; r < 8; ++r) {
    int om = m0 + ((wv >> 1) * 16) + r + ((lane & 16) ? 8 : 0);
    float inv  = bns[om] * rsqrtf(bnv[om] + 1e-5f);
    float beta = bnb[om] - bnm[om] * inv;
    float bi   = bias[om];
    for (int half = 0; half < 2; ++half) {
      int on = n0 + nsub + half * 16 + (lane & 15);
      int xc = on % W; int t2 = on / W;
      int y = t2 % H;  int bb = t2 / H;
      float v = ((half ? acc1[r] : acc0[r]) + bi) * inv + beta;
      out[((bb * O + om) * H + y) * W + xc] = (_Float16)fmaxf(v, 0.0f);
    }
  }
}

// ---------------------------------------------------------------------------
// Kernel 4: 2x2/2 max pool on f16 NCHW.
// ---------------------------------------------------------------------------
__global__ void maxpool2_kernel(const _Float16* __restrict__ in,
                                _Float16* __restrict__ out,
                                int C, int H, int W) {
  int Ho = H >> 1, Wo = W >> 1;
  long i = (long)blockIdx.x * blockDim.x + threadIdx.x;
  long total = 128L * C * Ho * Wo;
  if (i >= total) return;
  int ox = (int)(i % Wo); long t = i / Wo;
  int oy = (int)(t % Ho); t /= Ho;
  int c  = (int)(t % C);  int b = (int)(t / C);
  const _Float16* p = in + ((long)(b * C + c) * H + oy * 2) * W + ox * 2;
  float v0 = (float)p[0], v1 = (float)p[1];
  float v2 = (float)p[W], v3 = (float)p[W + 1];
  out[i] = (_Float16)fmaxf(fmaxf(v0, v1), fmaxf(v2, v3));
}

// ---------------------------------------------------------------------------
// Kernel 5: FC layer, LDS-tiled WMMA GEMM with fused qvector concat.
// fc1 B-tiles staged with async global->LDS b128 (rows 16B-aligned).
// ---------------------------------------------------------------------------
__global__ void fc_wmma(const _Float16* __restrict__ src, int srcStride,
                        const float* __restrict__ qv,
                        const float* __restrict__ w_,
                        const float* __restrict__ bias,
                        int OutF, int K,
                        void* __restrict__ outp, int relu, int out_f32) {
  __shared__ __align__(16) _Float16 Af[64 * 40];
  __shared__ __align__(16) _Float16 Bf[32 * 40];
  int tid = threadIdx.x, lane = tid & 31, wv = tid >> 5;
  int mb = blockIdx.x >> 2, nb = blockIdx.x & 3;
  int m0 = mb * 64, n0 = nb * 32;
  int ksteps = (K + 31) >> 5;
  int arow  = ((wv >> 1) * 16) + (lane & 15);
  int kc    = (lane & 16) ? 16 : 0;
  int brow  = ((wv & 1) * 16) + (lane & 15);
  int kboff = (lane & 16) ? 16 : 0;
#if HAS_ASYNC_LDS
  bool async_ok = ((srcStride & 7) == 0);
#endif
  v8f acc = {};
  for (int ks = 0; ks < ksteps; ++ks) {
    int k0 = ks << 5;
    for (int i = tid; i < 2048; i += 256) {        // A: weights -> f16
      int m = i >> 5, kk = i & 31, k = k0 + kk;
      float av = ((m0 + m) < OutF && k < K) ? w_[(long)(m0 + m) * K + k] : 0.0f;
      Af[m * 40 + kk] = (_Float16)av;
    }
#if HAS_ASYNC_LDS
    if (async_ok && k0 >= 64 && (k0 + 32) <= K) {
      if (tid < 128) {                             // 32 rows x 4 b128 chunks
        int n = tid >> 2, q = tid & 3;
        const _Float16* gp = src + (long)(n0 + n) * srcStride + (k0 - 64) + q * 8;
        __builtin_amdgcn_global_load_async_to_lds_b128(GAS_I4(gp),
                                                       LAS_I4(&Bf[n * 40 + q * 8]),
                                                       0, 0);
      }
      __builtin_amdgcn_s_wait_asynccnt(0);
    } else
#endif
    {
      for (int i = tid; i < 1024; i += 256) {      // B: concat(qv, src)
        int n = i >> 5, kk = i & 31, k = k0 + kk;
        int bb = n0 + n;
        _Float16 bv = (_Float16)0.0f;
        if (k < 64)     bv = (_Float16)qv[bb * 64 + k];
        else if (k < K) bv = src[(long)bb * srcStride + (k - 64)];
        Bf[n * 40 + kk] = bv;
      }
    }
    __syncthreads();
    v16h A = frag16(&Af[arow * 40], kc, kc + 8);
    v16h B = frag16(&Bf[brow * 40 + kboff], 0, 8);
    acc = __builtin_amdgcn_wmma_f32_16x16x32_f16(false, A, false, B,
                                                 (short)0, acc, false, false);
    __syncthreads();
  }
  for (int r = 0; r < 8; ++r) {
    int om = m0 + ((wv >> 1) * 16) + r + ((lane & 16) ? 8 : 0);
    int bb = n0 + ((wv & 1) * 16) + (lane & 15);
    if (om >= OutF) continue;
    float v = acc[r] + bias[om];
    if (relu) v = fmaxf(v, 0.0f);
    if (out_f32) ((float*)outp)[bb * OutF + om] = v;
    else         ((_Float16*)outp)[bb * OutF + om] = (_Float16)v;
  }
}

// ---------------------------------------------------------------------------
// Host orchestration.  R0/R1 ping-pong (132 MB each) + R2 for f16 weights.
// ---------------------------------------------------------------------------
extern "C" void kernel_launch(void* const* d_in, const int* in_sizes, int n_in,
                              void* d_out, int out_size, void* d_ws, size_t ws_size,
                              hipStream_t stream) {
  (void)in_sizes; (void)n_in; (void)out_size; (void)ws_size;
  const float* x     = (const float*)d_in[0];
  const float* qv    = (const float*)d_in[1];
  const float* basis = (const float*)d_in[2];
  const float* c1aw  = (const float*)d_in[3];  const float* c1ab = (const float*)d_in[4];
  const float* c1bw  = (const float*)d_in[5];  const float* c1bb = (const float*)d_in[6];
  const float* c2aw  = (const float*)d_in[7];  const float* c2ab = (const float*)d_in[8];
  const float* c3aw  = (const float*)d_in[9];  const float* c3ab = (const float*)d_in[10];
  const float* bn1a_s = (const float*)d_in[11]; const float* bn1a_b = (const float*)d_in[12];
  const float* bn1a_m = (const float*)d_in[13]; const float* bn1a_v = (const float*)d_in[14];
  const float* bn1b_s = (const float*)d_in[15]; const float* bn1b_b = (const float*)d_in[16];
  const float* bn1b_m = (const float*)d_in[17]; const float* bn1b_v = (const float*)d_in[18];
  const float* bn2a_s = (const float*)d_in[19]; const float* bn2a_b = (const float*)d_in[20];
  const float* bn2a_m = (const float*)d_in[21]; const float* bn2a_v = (const float*)d_in[22];
  const float* bn3a_s = (const float*)d_in[23]; const float* bn3a_b = (const float*)d_in[24];
  const float* bn3a_m = (const float*)d_in[25]; const float* bn3a_v = (const float*)d_in[26];
  const float* fc1w = (const float*)d_in[27]; const float* fc1b = (const float*)d_in[28];
  const float* fc2w = (const float*)d_in[29]; const float* fc2b = (const float*)d_in[30];
  const float* fc3w = (const float*)d_in[31]; const float* fc3b = (const float*)d_in[32];

  char* ws = (char*)d_ws;
  const size_t MB = 1ull << 20;
  char* R0 = ws;
  char* R1 = ws + 132 * MB;

  float*     dctb = (float*)R0;        // [128,64,1024] f32
  _Float16*  feat = (_Float16*)R1;     // [128,120,64]
  _Float16*  h1a  = (_Float16*)R0;     // [128,64,120,64]
  _Float16*  h1b  = (_Float16*)R1;
  _Float16*  p1   = (_Float16*)R0;     // [128,64,60,32]
  _Float16*  h2   = (_Float16*)R1;     // [128,128,60,32]
  _Float16*  p2   = (_Float16*)R0;     // [128,128,30,16]
  _Float16*  h3   = (_Float16*)R1;     // [128,256,30,16]
  _Float16*  p3   = (_Float16*)R0;     // [128,256,15,8]
  _Float16*  f1o  = (_Float16*)R1;     // [128,500]
  _Float16*  f2o  = (_Float16*)R0;     // [128,500]

  // f16 weight cache (2.3 MB), 16B-aligned slabs
  _Float16* w1a16 = (_Float16*)(ws + 264 * MB);   //   1600 elems (K=25)
  _Float16* w1b16 = w1a16 + 1600;                 // 102400 elems (K=1600)
  _Float16* w2a16 = w1b16 + 102400;               // 204800 elems (K=1600)
  _Float16* w3a16 = w2a16 + 204800;               // 819200 elems (K=3200)

  dim3 blk(256);

  cvt_f16_kernel<<<7,    blk, 0, stream>>>(c1aw, w1a16, 1600);
  cvt_f16_kernel<<<400,  blk, 0, stream>>>(c1bw, w1b16, 102400);
  cvt_f16_kernel<<<800,  blk, 0, stream>>>(c2aw, w2a16, 204800);
  cvt_f16_kernel<<<3200, blk, 0, stream>>>(c3aw, w3a16, 819200);

  dct_wmma_kernel<<<4096, blk, 0, stream>>>(x, basis, dctb);
  hist_kernel<<<8192, blk, 0, stream>>>(dctb, feat);

  // grids = (O/64) * (128*H*W/64)
  conv_bn_relu_wmma<<<15360, blk, 0, stream>>>(feat, w1a16, c1ab,
      bn1a_s, bn1a_b, bn1a_m, bn1a_v, h1a, 1, 64, 120, 64);
  conv_bn_relu_wmma<<<15360, blk, 0, stream>>>(h1a, w1b16, c1bb,
      bn1b_s, bn1b_b, bn1b_m, bn1b_v, h1b, 64, 64, 120, 64);
  maxpool2_kernel<<<61440, blk, 0, stream>>>(h1b, p1, 64, 120, 64);
  conv_bn_relu_wmma<<<7680, blk, 0, stream>>>(p1, w2a16, c2ab,
      bn2a_s, bn2a_b, bn2a_m, bn2a_v, h2, 64, 128, 60, 32);
  maxpool2_kernel<<<30720, blk, 0, stream>>>(h2, p2, 128, 60, 32);
  conv_bn_relu_wmma<<<3840, blk, 0, stream>>>(p2, w3a16, c3ab,
      bn3a_s, bn3a_b, bn3a_m, bn3a_v, h3, 128, 256, 30, 16);
  maxpool2_kernel<<<15360, blk, 0, stream>>>(h3, p3, 256, 30, 16);

  fc_wmma<<<32, blk, 0, stream>>>(p3, 30720, qv, fc1w, fc1b, 500, 30784, f1o, 1, 0);
  fc_wmma<<<32, blk, 0, stream>>>(f1o, 500, qv, fc2w, fc2b, 500, 564, f2o, 1, 0);
  fc_wmma<<<4,  blk, 0, stream>>>(f2o, 500, qv, fc3w, fc3b, 2,   564, d_out, 0, 1);
}
